// ToeplitzHeads_48137993453905
// MI455X (gfx1250) — matrix operand order repaired and checked by reading, not compiled, not run on hardware
//
#include <hip/hip_runtime.h>

// ToeplitzHeads on MI455X (gfx1250):
//   stage1: proj  = WpFlat(DxD) x x[b](DxT) + bp      (bf16 WMMA, f32 acc)
//   stage2: mixed = proj_h(ExT) x Toeplitz^T(TxT)     (causal, K-blocks skipped)
//   stage3: out   = Wo(DxD) x mixed[b](DxT) + bo      (f32 output)
// v_wmma_f32_16x16x32_bf16 GEMMs; A-tiles staged with CDNA5 async copies
// (global_load_async_to_lds_b128 + s_wait_asynccnt); double-buffered LDS,
// one barrier per K-step; branch-free Toeplitz B-tiles from a reversed,
// zero-padded generator row in LDS.

constexpr int Bc = 4, Dd = 1024, Tt = 2048, Hh = 16, Ee = 64;

typedef unsigned short u16x8  __attribute__((ext_vector_type(8)));
typedef __bf16         bf16x8 __attribute__((ext_vector_type(8)));
typedef __bf16         bf16x16 __attribute__((ext_vector_type(16)));
typedef float          f32x8  __attribute__((ext_vector_type(8)));

#define KP 40               // padded K stride (elems): 80B rows, 16B aligned
#define TILE_A (128 * KP)
#define TILE_B (128 * KP)
#define TILE_A2 (64 * KP)
#define REVPAD 160          // zero pad covering the causal d<0 region

__device__ __forceinline__ unsigned short f32_to_bf16(float f) {
    unsigned u = __float_as_uint(f);
    u += 0x7FFFu + ((u >> 16) & 1u);            // round-to-nearest-even
    return (unsigned short)(u >> 16);
}

// CDNA5 async copy: 16B global -> LDS, tracked by ASYNCcnt (no VGPR bounce).
__device__ __forceinline__ void async_b128(unsigned lds_off, const void* gptr) {
    asm volatile("global_load_async_to_lds_b128 %0, %1, off"
                 :: "v"(lds_off), "v"(gptr) : "memory");
}
__device__ __forceinline__ void wait_async0() {
    asm volatile("s_wait_asynccnt 0x0" ::: "memory");
}
// Low 32 bits of a flat shared-pointer == LDS byte offset (flat LDS mapping).
__device__ __forceinline__ unsigned lds_off(const void* p) {
    return (unsigned)(size_t)p;
}

// Build a v16bf fragment from two 16-byte LDS chunks (ds_load_b128 pairs).
__device__ __forceinline__ bf16x16 frag16(const unsigned short* p0,
                                          const unsigned short* p1) {
    bf16x8 lo = *(const bf16x8*)p0;
    bf16x8 hi = *(const bf16x8*)p1;
    return __builtin_shufflevector(lo, hi, 0,1,2,3,4,5,6,7,8,9,10,11,12,13,14,15);
}

__global__ void f2bf_kernel(const float* __restrict__ in,
                            unsigned short* __restrict__ out, int n) {
    int i = blockIdx.x * blockDim.x + threadIdx.x;
    if (i < n) out[i] = f32_to_bf16(in[i]);
}

// ---------------------------------------------------------------------------
// Stages 1 & 3: C[b](DxT) = A(DxD) x Bm[b](DxT) + bias, tile 128x128x32.
// ---------------------------------------------------------------------------
template<bool OUT_F32>
__global__ __launch_bounds__(256)
void gemm_dd_kernel(const unsigned short* __restrict__ A,
                    const unsigned short* __restrict__ Bm,
                    const float* __restrict__ bias,
                    void* __restrict__ Cout) {
    __shared__ unsigned short Al[2 * TILE_A];
    __shared__ unsigned short Bl[2 * TILE_B];

    const int tid  = threadIdx.x;
    const int lane = tid & 31;
    const int wid  = tid >> 5;
    const int hl   = lane >> 4;
    const int ln   = lane & 15;

    const int m0 = blockIdx.y * 128;
    const int n0 = blockIdx.x * 128;
    const int b  = blockIdx.z;

    const unsigned short* Bg = Bm + (size_t)b * Dd * Tt;

    const int mw = (wid & 1) * 64;   // wave tile: 64(M) x 32(N)
    const int nw = (wid >> 1) * 32;

    f32x8 acc[4][2] = {};

    const int a_m  = tid >> 1;             // 0..127
    const int a_k  = (tid & 1) * 16;       // 0 / 16
    const int b_n8 = (tid & 15) * 8;       // 0..120  (coalesced n)
    const int b_k2 = (tid >> 4) * 2;       // 0..30   (even k pair)

    u16x8 bv0, bv1;                        // B prefetch registers (rows k, k+1)

    auto issueA = [&](int k0, int buf) {   // async: global -> LDS, no regs
        const unsigned short* ag = A + (size_t)(m0 + a_m) * Dd + k0 + a_k;
        unsigned l0 = lds_off(&Al[buf * TILE_A + a_m * KP + a_k]);
        async_b128(l0,      ag);
        async_b128(l0 + 16, ag + 8);
    };
    auto fetchB = [&](int k0) {
        const unsigned short* r0 = Bg + (size_t)(k0 + b_k2) * Tt + n0 + b_n8;
        bv0 = *(const u16x8*)r0;
        bv1 = *(const u16x8*)(r0 + Tt);
    };
    auto commitB = [&](int buf) {          // transpose-on-store, dword-packed
        unsigned short* Bb = &Bl[buf * TILE_B];
        #pragma unroll
        for (int i = 0; i < 8; ++i) {
            unsigned pk = (unsigned)bv0[i] | ((unsigned)bv1[i] << 16);
            *(unsigned*)&Bb[(b_n8 + i) * KP + b_k2] = pk;
        }
    };

    issueA(0, 0); fetchB(0); commitB(0);
    wait_async0();
    __syncthreads();

    int buf = 0;
    for (int k0 = 0; k0 < Dd; k0 += 32, buf ^= 1) {
        const bool more = (k0 + 32) < Dd;
        if (more) { issueA(k0 + 32, buf ^ 1); fetchB(k0 + 32); }

        const unsigned short* Ab = &Al[buf * TILE_A];
        const unsigned short* Bb = &Bl[buf * TILE_B];
        bf16x16 af[4], bfv[2];
        #pragma unroll
        for (int i = 0; i < 4; ++i) {
            const unsigned short* r = &Ab[(mw + i * 16 + ln) * KP];
            af[i] = frag16(r + 8 * hl, r + 16 + 8 * hl);
        }
        #pragma unroll
        for (int j = 0; j < 2; ++j) {
            const unsigned short* r = &Bb[(nw + j * 16 + ln) * KP + 8 * hl];
            bfv[j] = frag16(r, r + 16);
        }
        #pragma unroll
        for (int i = 0; i < 4; ++i)
            #pragma unroll
            for (int j = 0; j < 2; ++j)
                acc[i][j] = __builtin_amdgcn_wmma_f32_16x16x32_bf16(
                    false, af[i], false, bfv[j], (short)0, acc[i][j], false, false);

        if (more) commitB(buf ^ 1);
        wait_async0();
        __syncthreads();
    }

    // Epilogue: C VGPR v, lane l -> (M = v + 8*hl, N = l%16).
    #pragma unroll
    for (int i = 0; i < 4; ++i)
        #pragma unroll
        for (int j = 0; j < 2; ++j)
            #pragma unroll
            for (int v = 0; v < 8; ++v) {
                int mg = m0 + mw + i * 16 + v + 8 * hl;
                int ng = n0 + nw + j * 16 + ln;
                float val = acc[i][j][v] + bias[mg];
                size_t idx = (size_t)b * Dd * Tt + (size_t)mg * Tt + ng;
                if (OUT_F32) ((float*)Cout)[idx] = val;
                else ((unsigned short*)Cout)[idx] = f32_to_bf16(val);
            }
}

// ---------------------------------------------------------------------------
// Stage 2: per (b,h): mixed(64xT) = proj_h(64xT) x Toep^T, Toep^T[s,t]=tw[t-s].
// Tile 64x128x32. Rev[u] = u<T ? tw[T-1-u] : 0 makes the Toeplitz B-tile
// build 16 contiguous unconditional LDS reads. Causal K-blocks only.
// ---------------------------------------------------------------------------
__global__ __launch_bounds__(256)
void toeplitz_mix_kernel(const unsigned short* __restrict__ P,
                         const unsigned short* __restrict__ twg,
                         unsigned short* __restrict__ Mx) {
    __shared__ unsigned short Al[2 * TILE_A2];
    __shared__ unsigned short Bl[2 * TILE_B];
    __shared__ unsigned short Rev[Tt + REVPAD];

    const int tid  = threadIdx.x;
    const int lane = tid & 31;
    const int wid  = tid >> 5;
    const int hl   = lane >> 4;
    const int ln   = lane & 15;

    const int t0 = blockIdx.x * 128;
    const int h  = blockIdx.y;
    const int b  = blockIdx.z;

    for (int u = tid; u < Tt + REVPAD; u += 256)
        Rev[u] = (u < Tt) ? twg[(size_t)h * Tt + (Tt - 1 - u)] : (unsigned short)0;
    __syncthreads();

    const unsigned short* Pg = P + (size_t)b * Dd * Tt + (size_t)(h * Ee) * Tt;

    const int mw = (wid & 1) * 32;   // wave tile: 32(M) x 32(N)
    const int nw = (wid >> 1) * 32;

    f32x8 acc[2][2] = {};

    const int a_m  = tid >> 2;           // 0..63
    const int a_k  = (tid & 3) * 8;      // 0..24
    const int b_n  = tid >> 1;           // 0..127
    const int b_kh = (tid & 1) * 16;     // 0 / 16

    const int ksteps = (t0 >> 5) + 4;    // covers s in [0, t0+128)

    u16x8 br0, br1;
    auto issueA = [&](int ks, int buf) {
        async_b128(lds_off(&Al[buf * TILE_A2 + a_m * KP + a_k]),
                   Pg + (size_t)a_m * Tt + ks * 32 + a_k);
    };
    auto fetchB = [&](int ks) {
        const int u0 = (Tt - 1) - (t0 + b_n) + ks * 32 + b_kh;   // >= 0 always
        #pragma unroll
        for (int i = 0; i < 8; ++i) br0[i] = Rev[u0 + i];
        #pragma unroll
        for (int i = 0; i < 8; ++i) br1[i] = Rev[u0 + 8 + i];
    };
    auto commitB = [&](int buf) {
        *(u16x8*)&Bl[buf * TILE_B + b_n * KP + b_kh]     = br0;
        *(u16x8*)&Bl[buf * TILE_B + b_n * KP + b_kh + 8] = br1;
    };

    issueA(0, 0); fetchB(0); commitB(0);
    wait_async0();
    __syncthreads();

    int buf = 0;
    for (int ks = 0; ks < ksteps; ++ks, buf ^= 1) {
        const bool more = (ks + 1) < ksteps;
        if (more) { issueA(ks + 1, buf ^ 1); fetchB(ks + 1); }

        const unsigned short* Ab = &Al[buf * TILE_A2];
        const unsigned short* Bb = &Bl[buf * TILE_B];
        bf16x16 af[2], bfv[2];
        #pragma unroll
        for (int i = 0; i < 2; ++i) {
            const unsigned short* r = &Ab[(mw + i * 16 + ln) * KP];
            af[i] = frag16(r + 8 * hl, r + 16 + 8 * hl);
        }
        #pragma unroll
        for (int j = 0; j < 2; ++j) {
            const unsigned short* r = &Bb[(nw + j * 16 + ln) * KP + 8 * hl];
            bfv[j] = frag16(r, r + 16);
        }
        #pragma unroll
        for (int i = 0; i < 2; ++i)
            #pragma unroll
            for (int j = 0; j < 2; ++j)
                acc[i][j] = __builtin_amdgcn_wmma_f32_16x16x32_bf16(
                    false, af[i], false, bfv[j], (short)0, acc[i][j], false, false);

        if (more) commitB(buf ^ 1);
        wait_async0();
        __syncthreads();
    }

    #pragma unroll
    for (int i = 0; i < 2; ++i)
        #pragma unroll
        for (int j = 0; j < 2; ++j)
            #pragma unroll
            for (int v = 0; v < 8; ++v) {
                int mg = h * Ee + mw + i * 16 + v + 8 * hl;
                int ng = t0 + nw + j * 16 + ln;
                Mx[(size_t)b * Dd * Tt + (size_t)mg * Tt + ng] =
                    f32_to_bf16(acc[i][j][v]);
            }
}

extern "C" void kernel_launch(void* const* d_in, const int* in_sizes, int n_in,
                              void* d_out, int out_size, void* d_ws, size_t ws_size,
                              hipStream_t stream) {
    const float* x  = (const float*)d_in[0];   // (B, D, T)
    const float* Wp = (const float*)d_in[1];   // (H, E, D) == (D, D) flat
    const float* bp = (const float*)d_in[2];   // (H, E)    == (D,)  flat
    const float* tw = (const float*)d_in[3];   // (H, T)
    const float* Wo = (const float*)d_in[4];   // (D, D)
    const float* bo = (const float*)d_in[5];   // (D,)

    char* ws = (char*)d_ws;
    size_t off = 0;
    unsigned short* xbf    = (unsigned short*)(ws + off); off += (size_t)Bc*Dd*Tt*2;
    unsigned short* wpbf   = (unsigned short*)(ws + off); off += (size_t)Dd*Dd*2;
    unsigned short* twbf   = (unsigned short*)(ws + off); off += (size_t)Hh*Tt*2;
    unsigned short* wobf   = (unsigned short*)(ws + off); off += (size_t)Dd*Dd*2;
    unsigned short* projbf = (unsigned short*)(ws + off); off += (size_t)Bc*Dd*Tt*2;
    unsigned short* mixbf  = (unsigned short*)(ws + off); off += (size_t)Bc*Dd*Tt*2;

    const int nX = Bc * Dd * Tt, nW = Dd * Dd, nTw = Hh * Tt;
    f2bf_kernel<<<(nX  + 255) / 256, 256, 0, stream>>>(x,  xbf,  nX);
    f2bf_kernel<<<(nW  + 255) / 256, 256, 0, stream>>>(Wp, wpbf, nW);
    f2bf_kernel<<<(nTw + 255) / 256, 256, 0, stream>>>(tw, twbf, nTw);
    f2bf_kernel<<<(nW  + 255) / 256, 256, 0, stream>>>(Wo, wobf, nW);

    dim3 g13(Tt / 128, Dd / 128, Bc);   // 16 x 8 x 4
    dim3 g2 (Tt / 128, Hh,       Bc);   // 16 x 16 x 4

    gemm_dd_kernel<false><<<g13, 256, 0, stream>>>(wpbf, xbf, bp, (void*)projbf);
    toeplitz_mix_kernel  <<<g2,  256, 0, stream>>>(projbf, twbf, mixbf);
    gemm_dd_kernel<true> <<<g13, 256, 0, stream>>>(wobf, mixbf, bo, d_out);
}
